// MultiHeadAttention_40948218200338
// MI455X (gfx1250) — matrix operand order repaired and checked by reading, not compiled, not run on hardware
//
#include <hip/hip_runtime.h>
#include <hip/hip_bf16.h>

#define Bsz 2
#define Ssz 2048
#define Dsz 1024
#define Hsz 16
#define DKsz 64

typedef __attribute__((ext_vector_type(16))) __bf16 v16bf;
typedef __attribute__((ext_vector_type(2)))  __bf16 v2bf;
typedef __attribute__((ext_vector_type(8)))  float  v8f;

union BFrag {
    unsigned int u[8];
    uint4        q[2];
    v16bf        v;
};

// native bf16 conversion -> backend picks v_cvt_*bf16* hardware path
static __device__ __forceinline__ unsigned short f2bf(float f) {
    union { __bf16 b; unsigned short s; } t;
    t.b = (__bf16)f;
    return t.s;
}
static __device__ __forceinline__ unsigned int pack2(float lo, float hi) {
    union { v2bf v; unsigned int u; } t;
    t.v[0] = (__bf16)lo;
    t.v[1] = (__bf16)hi;
    return t.u;
}

// ---------------------------------------------------------------------------
// One-time fp32 -> bf16 conversion (weights). Bandwidth-bound, runs once.
// ---------------------------------------------------------------------------
__global__ __launch_bounds__(256) void mha_cvt_bf16_kernel(
    const float* __restrict__ in, unsigned short* __restrict__ out, int n4)
{
    const int i = blockIdx.x * 256 + threadIdx.x;
    if (i < n4) {
        const float4 f = ((const float4*)in)[i];
        uint2 p;
        p.x = pack2(f.x, f.y);
        p.y = pack2(f.z, f.w);
        ((uint2*)out)[i] = p;
    }
}

// ---------------------------------------------------------------------------
// GEMM: Y = X(fp32) @ Wb^T(bf16) + bias.  A tile (16x1024) is converted to
// bf16 into LDS once per block; 8 waves each do one 16x16 output tile via
// 32x v_wmma_f32_16x16x32_bf16 reading A from LDS (ds_load_b128) and B from
// global bf16 (global_load_b128).
//   transposeV==0 : store (B,H,S,DK)   (Q, K)
//   transposeV==1 : store (B,H,DK,S)   (V)
// ---------------------------------------------------------------------------
__global__ __launch_bounds__(256) void mha_proj_kernel(
    const float* __restrict__ X, const unsigned short* __restrict__ Wb,
    const float* __restrict__ bias, unsigned short* __restrict__ out,
    float scale, int transposeV)
{
    __shared__ unsigned short atile[16 * Dsz];          // 32 KB bf16 A tile
    const int tid  = threadIdx.x;
    const int row0 = blockIdx.y * 16;

    // cooperative stage: 16 rows x 1024 cols fp32 -> bf16 LDS
    {
        const float4* src = (const float4*)(X + (size_t)row0 * Dsz);
        uint2* dst = (uint2*)atile;
#pragma unroll
        for (int i = 0; i < 16; ++i) {                  // 4096 float4 / 256 thr
            const int idx = i * 256 + tid;
            const float4 f = src[idx];
            uint2 p;
            p.x = pack2(f.x, f.y);
            p.y = pack2(f.z, f.w);
            dst[idx] = p;
        }
    }
    __syncthreads();

    const int lane = tid & 31;
    const int wave = tid >> 5;
    const int lr   = lane & 15;
    const int sub  = lane >> 4;
    const int col0 = (blockIdx.x * 8 + wave) * 16;

    const unsigned short* arow = atile + lr * Dsz;                 // LDS
    const unsigned short* brow = Wb + (size_t)(col0 + lr) * Dsz;   // global bf16

    v8f acc = {};
    for (int k0 = 0; k0 < Dsz; k0 += 32) {
        BFrag a, b;
        // A 16x32: v -> K = 16*sub + 2v
        a.q[0] = *(const uint4*)(arow + k0 + 16 * sub);
        a.q[1] = *(const uint4*)(arow + k0 + 16 * sub + 8);
        // B 32x16: v -> K = 16*sub + 2v
        b.q[0] = *(const uint4*)(brow + k0 + 16 * sub);
        b.q[1] = *(const uint4*)(brow + k0 + 16 * sub + 8);
        __builtin_prefetch(brow + k0 + 64, 0, 3);        // global_prefetch_b8
        acc = __builtin_amdgcn_wmma_f32_16x16x32_bf16(
            false, a.v, false, b.v, (short)0, acc, false, false);
    }

    const int   n    = col0 + lr;
    const float bval = bias[n];
    const int   h    = n >> 6;
    const int   dk   = n & 63;
#pragma unroll
    for (int r = 0; r < 8; ++r) {
        const int grow = row0 + r + 8 * sub;             // [0, B*S)
        const int bb   = grow >> 11;                     // / Ssz
        const int s    = grow & (Ssz - 1);
        const float v  = (acc[r] + bval) * scale;
        size_t idx;
        if (transposeV)
            idx = (((size_t)bb * Hsz + h) * DKsz + dk) * Ssz + s;
        else
            idx = (((size_t)bb * Hsz + h) * Ssz + s) * DKsz + dk;
        out[idx] = f2bf(v);
    }
}

// ---------------------------------------------------------------------------
// Flash-style causal attention. One wave owns a 16-row q-tile x DK=64 output.
// Streams 32-wide k-tiles: scores via 4 WMMAs, online softmax, P bounced
// through LDS (C-layout -> A-fragment), then P@V via 4 WMMAs.
// Q pre-scaled by 1/sqrt(DK). Output: bf16 context, (B,S,D) row-major.
// ---------------------------------------------------------------------------
__global__ __launch_bounds__(256) void mha_attn_kernel(
    const unsigned short* __restrict__ Q, const unsigned short* __restrict__ K,
    const unsigned short* __restrict__ Vt, unsigned short* __restrict__ Ctx)
{
    __shared__ unsigned short ptile[8 * 16 * 32];        // 1 KB per wave
    const int lane = threadIdx.x & 31;
    const int wave = threadIdx.x >> 5;
    const int lr   = lane & 15;
    const int sub  = lane >> 4;
    const int q0   = (blockIdx.x * 8 + wave) * 16;
    const int bh   = blockIdx.y;

    const unsigned short* Qb = Q  + (size_t)bh * Ssz * DKsz;
    const unsigned short* Kb = K  + (size_t)bh * Ssz * DKsz;
    const unsigned short* Vb = Vt + (size_t)bh * DKsz * Ssz;
    unsigned short* pbuf = ptile + wave * (16 * 32);

    BFrag qa0, qa1;                                      // Q A-fragments, resident
    {
        const unsigned short* qrow = Qb + (size_t)(q0 + lr) * DKsz;
        qa0.q[0] = *(const uint4*)(qrow + 16 * sub);
        qa0.q[1] = *(const uint4*)(qrow + 16 * sub + 8);
        qa1.q[0] = *(const uint4*)(qrow + 32 + 16 * sub);
        qa1.q[1] = *(const uint4*)(qrow + 32 + 16 * sub + 8);
    }

    v8f o[4] = {};
    float m[8], l[8];
#pragma unroll
    for (int r = 0; r < 8; ++r) { m[r] = -1e30f; l[r] = 0.f; }

    const int kend = q0 + 16;                            // causal bound
    for (int kt = 0; kt < kend; kt += 32) {
        v8f s0 = {}, s1 = {};
        {
            int colg = kt + lr; if (colg > Ssz - 1) colg = Ssz - 1;
            const unsigned short* krow = Kb + (size_t)colg * DKsz;
            BFrag b0, b1;
            b0.q[0] = *(const uint4*)(krow + 16 * sub);
            b0.q[1] = *(const uint4*)(krow + 16 * sub + 8);
            b1.q[0] = *(const uint4*)(krow + 32 + 16 * sub);
            b1.q[1] = *(const uint4*)(krow + 32 + 16 * sub + 8);
            s0 = __builtin_amdgcn_wmma_f32_16x16x32_bf16(false, qa0.v, false, b0.v, (short)0, s0, false, false);
            s0 = __builtin_amdgcn_wmma_f32_16x16x32_bf16(false, qa1.v, false, b1.v, (short)0, s0, false, false);
        }
        {
            int colg = kt + 16 + lr; if (colg > Ssz - 1) colg = Ssz - 1;
            const unsigned short* krow = Kb + (size_t)colg * DKsz;
            BFrag b0, b1;
            b0.q[0] = *(const uint4*)(krow + 16 * sub);
            b0.q[1] = *(const uint4*)(krow + 16 * sub + 8);
            b1.q[0] = *(const uint4*)(krow + 32 + 16 * sub);
            b1.q[1] = *(const uint4*)(krow + 32 + 16 * sub + 8);
            s1 = __builtin_amdgcn_wmma_f32_16x16x32_bf16(false, qa0.v, false, b0.v, (short)0, s1, false, false);
            s1 = __builtin_amdgcn_wmma_f32_16x16x32_bf16(false, qa1.v, false, b1.v, (short)0, s1, false, false);
        }
        // causal mask (C layout: N = lr, M = r + 8*sub)
#pragma unroll
        for (int r = 0; r < 8; ++r) {
            const int mrow = q0 + r + 8 * sub;
            s0[r] = (kt + lr      <= mrow) ? s0[r] : -1e9f;
            s1[r] = (kt + 16 + lr <= mrow) ? s1[r] : -1e9f;
        }
        // online softmax: reduce across the 16 lanes of this half
        float t[8];
#pragma unroll
        for (int r = 0; r < 8; ++r) t[r] = fmaxf(s0[r], s1[r]);
#pragma unroll
        for (int off = 8; off; off >>= 1)
#pragma unroll
            for (int r = 0; r < 8; ++r) t[r] = fmaxf(t[r], __shfl_xor(t[r], off, 32));
        float alpha[8];
#pragma unroll
        for (int r = 0; r < 8; ++r) {
            const float nm = fmaxf(m[r], t[r]);
            alpha[r] = __expf(m[r] - nm);
            m[r] = nm;
        }
#pragma unroll
        for (int r = 0; r < 8; ++r) { s0[r] = __expf(s0[r] - m[r]); s1[r] = __expf(s1[r] - m[r]); }
#pragma unroll
        for (int r = 0; r < 8; ++r) t[r] = s0[r] + s1[r];
#pragma unroll
        for (int off = 8; off; off >>= 1)
#pragma unroll
            for (int r = 0; r < 8; ++r) t[r] += __shfl_xor(t[r], off, 32);
#pragma unroll
        for (int r = 0; r < 8; ++r) l[r] = l[r] * alpha[r] + t[r];
#pragma unroll
        for (int n = 0; n < 4; ++n)
#pragma unroll
            for (int r = 0; r < 8; ++r) o[n][r] *= alpha[r];

        // P (f32, C-layout) -> bf16 row-major LDS -> A-fragment
#pragma unroll
        for (int r = 0; r < 8; ++r) {
            const int row = r + 8 * sub;
            pbuf[row * 32 + lr]      = f2bf(s0[r]);
            pbuf[row * 32 + 16 + lr] = f2bf(s1[r]);
        }
        asm volatile("s_wait_dscnt 0" ::: "memory");
        BFrag pa;
        pa.q[0] = *(const uint4*)(pbuf + lr * 32 + 16 * sub);
        pa.q[1] = *(const uint4*)(pbuf + lr * 32 + 16 * sub + 8);

        // O += P @ V  (V transposed: Vb[d*S + k], contiguous in k)
        int kb = kt + 16 * sub; if (kb > Ssz - 16) kb = Ssz - 16;
#pragma unroll
        for (int n = 0; n < 4; ++n) {
            const unsigned short* vrow = Vb + (size_t)(n * 16 + lr) * Ssz + kb;
            BFrag vbf;
            vbf.q[0] = *(const uint4*)(vrow);
            vbf.q[1] = *(const uint4*)(vrow + 8);
            o[n] = __builtin_amdgcn_wmma_f32_16x16x32_bf16(
                false, pa.v, false, vbf.v, (short)0, o[n], false, false);
        }
    }

    const int b = bh >> 4;
    const int h = bh & 15;
#pragma unroll
    for (int r = 0; r < 8; ++r) {
        const float inv = 1.0f / l[r];
        const size_t grow = (size_t)b * Ssz + q0 + r + 8 * sub;
#pragma unroll
        for (int n = 0; n < 4; ++n) {
            const int col = h * DKsz + n * 16 + lr;
            Ctx[grow * Dsz + col] = f2bf(o[n][r] * inv);
        }
    }
}

// ---------------------------------------------------------------------------
// Output projection: out = Ctx(bf16) @ Wob^T(bf16) + bo, fp32 output.
// A tile copied to LDS once per block (ds_load_b128 fragments).
// ---------------------------------------------------------------------------
__global__ __launch_bounds__(256) void mha_oproj_kernel(
    const unsigned short* __restrict__ Ctx, const unsigned short* __restrict__ Wb,
    const float* __restrict__ bias, float* __restrict__ out)
{
    __shared__ unsigned short atile[16 * Dsz];           // 32 KB
    const int tid  = threadIdx.x;
    const int row0 = blockIdx.y * 16;
    {
        const uint4* src = (const uint4*)(Ctx + (size_t)row0 * Dsz);
        uint4* dst = (uint4*)atile;
#pragma unroll
        for (int i = 0; i < 8; ++i)                      // 2048 uint4 / 256 thr
            dst[i * 256 + tid] = src[i * 256 + tid];
    }
    __syncthreads();

    const int lane = tid & 31;
    const int wave = tid >> 5;
    const int lr   = lane & 15;
    const int sub  = lane >> 4;
    const int col0 = (blockIdx.x * 8 + wave) * 16;

    const unsigned short* arow = atile + lr * Dsz;
    const unsigned short* brow = Wb + (size_t)(col0 + lr) * Dsz;

    v8f acc = {};
    for (int k0 = 0; k0 < Dsz; k0 += 32) {
        BFrag a, b;
        a.q[0] = *(const uint4*)(arow + k0 + 16 * sub);
        a.q[1] = *(const uint4*)(arow + k0 + 16 * sub + 8);
        b.q[0] = *(const uint4*)(brow + k0 + 16 * sub);
        b.q[1] = *(const uint4*)(brow + k0 + 16 * sub + 8);
        __builtin_prefetch(brow + k0 + 64, 0, 3);
        acc = __builtin_amdgcn_wmma_f32_16x16x32_bf16(
            false, a.v, false, b.v, (short)0, acc, false, false);
    }

    const float bval = bias[col0 + lr];
#pragma unroll
    for (int r = 0; r < 8; ++r) {
        const size_t grow = row0 + r + 8 * sub;
        out[grow * Dsz + (col0 + lr)] = acc[r] + bval;
    }
}

// ---------------------------------------------------------------------------
extern "C" void kernel_launch(void* const* d_in, const int* in_sizes, int n_in,
                              void* d_out, int out_size, void* d_ws, size_t ws_size,
                              hipStream_t stream) {
    const float* query = (const float*)d_in[0];
    const float* key   = (const float*)d_in[1];
    const float* value = (const float*)d_in[2];
    // d_in[3] = causal mask, handled analytically in-kernel
    const float* Wq = (const float*)d_in[4];  const float* bq = (const float*)d_in[5];
    const float* Wk = (const float*)d_in[6];  const float* bk = (const float*)d_in[7];
    const float* Wv = (const float*)d_in[8];  const float* bv = (const float*)d_in[9];
    const float* Wo = (const float*)d_in[10]; const float* bo = (const float*)d_in[11];
    float* out = (float*)d_out;

    const size_t act = (size_t)Bsz * Ssz * Dsz;          // 4M elements
    const size_t wgt = (size_t)Dsz * Dsz;                // 1M elements
    unsigned short* Qws = (unsigned short*)d_ws;         // bf16 buffers:
    unsigned short* Kws = Qws + act;
    unsigned short* Vtw = Kws + act;
    unsigned short* Ctx = Vtw + act;
    unsigned short* Wqb = Ctx + act;
    unsigned short* Wkb = Wqb + wgt;
    unsigned short* Wvb = Wkb + wgt;
    unsigned short* Wob = Wvb + wgt;                     // total 40 MB

    dim3 blk(256);
    const int wn4 = (int)(wgt / 4);
    dim3 gcvt((wn4 + 255) / 256);
    mha_cvt_bf16_kernel<<<gcvt, blk, 0, stream>>>(Wq, Wqb, wn4);
    mha_cvt_bf16_kernel<<<gcvt, blk, 0, stream>>>(Wk, Wkb, wn4);
    mha_cvt_bf16_kernel<<<gcvt, blk, 0, stream>>>(Wv, Wvb, wn4);
    mha_cvt_bf16_kernel<<<gcvt, blk, 0, stream>>>(Wo, Wob, wn4);

    dim3 gproj(Dsz / 128, (Bsz * Ssz) / 16);             // (8, 256)
    dim3 gattn(Ssz / 128, Bsz * Hsz);                    // (16, 32)

    const float qscale = 0.125f;                         // 1/sqrt(DK)
    mha_proj_kernel<<<gproj, blk, 0, stream>>>(query, Wqb, bq, Qws, qscale, 0);
    mha_proj_kernel<<<gproj, blk, 0, stream>>>(key,   Wkb, bk, Kws, 1.0f,   0);
    mha_proj_kernel<<<gproj, blk, 0, stream>>>(value, Wvb, bv, Vtw, 1.0f,   1);
    mha_attn_kernel<<<gattn, blk, 0, stream>>>(Qws, Kws, Vtw, Ctx);
    mha_oproj_kernel<<<gproj, blk, 0, stream>>>(Ctx, Wob, bo, out);
}
